// Greedy_85040352460908
// MI455X (gfx1250) — compile-verified, matrix operand order checked by reading, not run on hardware
//
#include <hip/hip_runtime.h>
#include <stdint.h>

#define L 2048
#define LT 64            // TDM tile edge
#define NROUNDS 64       // locally-dominant matching rounds (expected O(log L) needed)

typedef __attribute__((ext_vector_type(16))) _Float16 v16h;
typedef __attribute__((ext_vector_type(8)))  float    v8f;
typedef __attribute__((ext_vector_type(4)))  unsigned int v4u;
typedef __attribute__((ext_vector_type(8)))  int      v8i;
typedef __attribute__((ext_vector_type(4)))  int      v4i;

// ---------- Kernel A: per-position base index u[i]=argmax_c feat[0,c,i,0], degen flag ----------
__global__ void k_bases(const float* __restrict__ feat, int* __restrict__ u,
                        int* __restrict__ degen) {
  int i = blockIdx.x * blockDim.x + threadIdx.x;
  if (i >= L) return;
  float best = -1.0f; int bi = 0;
  #pragma unroll
  for (int c = 0; c < 4; ++c) {
    float v = feat[(size_t)c * L * L + (size_t)i * L];   // feat[0,c,i,0]
    if (v > best) { best = v; bi = c; }                  // first-max wins (matches jnp.argmax)
  }
  u[i] = bi;
  degen[i] = (best < 1.0f) ? 1 : 0;
}

// ---------- Kernel B: build masked symmetric score matrix X (written into d_out) ----------
// One block = one 64x64 output tile. TDM stages the transposed-source tile into LDS
// with a 1-dword row pad (65-float pitch -> conflict-free transposed reads).
// Each of the 4 waves owns a 16-wide column strip; compat mask per 16x16 subtile via
// v_wmma_f32_16x16x32_f16:  D = A(16x32) x B(32x16),  A[m,k]=M[u[r]][k] (k<4), B[k,n]=[k==u[c]].
__global__ __launch_bounds__(128) void k_build(const float* __restrict__ con,
                                               const int* __restrict__ u,
                                               const int* __restrict__ degen,
                                               float* __restrict__ X) {
  __shared__ float ldsT[LT * (LT + 1)];   // TDM pad gives pitch 65 floats
  __shared__ int uR[LT], dR[LT], uC[LT], dC[LT];

  const int nb = L / LT;
  const int r0 = (blockIdx.x / nb) * LT;
  const int c0 = (blockIdx.x % nb) * LT;

  // Stage per-tile metadata into LDS (no conditional global loads in the hot loop).
  {
    const int t = threadIdx.x;
    if (t < LT)             { uR[t] = u[r0 + t];           dR[t] = degen[r0 + t]; }
    else if (t < 2 * LT)    { uC[t - LT] = u[c0 + t - LT]; dC[t - LT] = degen[c0 + t - LT]; }
  }

  // ---- TDM async tensor load: rows c0..c0+63, cols r0..r0+63 of con -> LDS ----
  if (threadIdx.x == 0) {
    unsigned long long gaddr = (unsigned long long)(uintptr_t)con
                             + ((unsigned long long)c0 * L + (unsigned long long)r0) * 4ull;
    unsigned int ldsb = (unsigned int)(uintptr_t)&ldsT[0]; // LDS byte offset of tile
    v4u g0;
    g0[0] = 1u;                                            // count = 1 (valid descriptor)
    g0[1] = ldsb;                                          // lds_addr
    g0[2] = (unsigned int)(gaddr & 0xFFFFFFFFull);         // global_addr[31:0]
    g0[3] = (unsigned int)((gaddr >> 32) & 0x1FFFFFFull) | (2u << 30); // addr[56:32] | type=2
    v8i g1;
    g1[0] = (2 << 16) | (1 << 20) | (5 << 22);             // data_size=4B, pad_enable, pad_interval=64DW (amount=1DW)
    g1[1] = (int)((L & 0xFFFF) << 16);                     // tensor_dim0 = 2048 (bits 79:48)
    g1[2] = (int)((L & 0xFFFF) << 16);                     // tensor_dim1 = 2048 (bits 111:80)
    g1[3] = (LT << 16);                                    // tile_dim0 = 64 (bits 127:112)
    g1[4] = LT;                                            // tile_dim1 = 64
    g1[5] = L;                                             // tensor_dim0_stride = 2048
    g1[6] = 0;
    g1[7] = 0;
    v4i gz = {0, 0, 0, 0};
#if defined(__clang_major__) && __clang_major__ >= 23
    v8i gz8 = {0, 0, 0, 0, 0, 0, 0, 0};
    __builtin_amdgcn_tensor_load_to_lds(g0, g1, gz, gz, gz8, 0);
#else
    __builtin_amdgcn_tensor_load_to_lds(g0, g1, gz, gz, 0);
#endif
    __builtin_amdgcn_s_wait_tensorcnt(0);
  }
  __syncthreads();

  const int lane = threadIdx.x & 31;
  const int wave = threadIdx.x >> 5;     // 0..3 -> 16-wide column strip
  const int hlf  = lane >> 4;
  const int l16  = lane & 15;

  // B operand: one-hot(u[c]) in K=0..3, lanes 0-15 only (K=0..7 slot), rest zero.
  const int cOff = wave * 16 + l16;
  const int uc = (hlf == 0) ? uC[cOff] : -1;
  v16h bmat = {0,0,0,0,0,0,0,0,0,0,0,0,0,0,0,0};
  bmat[0] = (_Float16)((uc == 0) ? 1.0f : 0.0f);
  bmat[1] = (_Float16)((uc == 1) ? 1.0f : 0.0f);
  bmat[2] = (_Float16)((uc == 2) ? 1.0f : 0.0f);
  bmat[3] = (_Float16)((uc == 3) ? 1.0f : 0.0f);

  const int cc = c0 + cOff;              // this lane's output column (C/D layout: n = lane&15)
  const int dc = dC[cOff];

  #pragma unroll
  for (int i = 0; i < 4; ++i) {
    const int rOff0 = i * 16;
    const int ur = (hlf == 0) ? uR[rOff0 + l16] : -1;
    // Row u[r] of compat table M (primes {2,3,5,7}; products 14,15,35):
    // M[.,0]=1 iff u==3 ; M[.,1]=1 iff u==2 ; M[.,2]=1 iff u in {1,3} ; M[.,3]=1 iff u in {0,2}
    v16h amat = {0,0,0,0,0,0,0,0,0,0,0,0,0,0,0,0};
    amat[0] = (_Float16)((ur == 3) ? 1.0f : 0.0f);
    amat[1] = (_Float16)((ur == 2) ? 1.0f : 0.0f);
    amat[2] = (_Float16)(((ur == 1) || (ur == 3)) ? 1.0f : 0.0f);
    amat[3] = (_Float16)(((ur == 0) || (ur == 2)) ? 1.0f : 0.0f);

    v8f cz = {0,0,0,0,0,0,0,0};
    v8f compat = __builtin_amdgcn_wmma_f32_16x16x32_f16(
        /*neg_a=*/false, amat, /*neg_b=*/false, bmat,
        /*c_mod=*/(short)0, cz, /*reuse_a=*/false, /*reuse_b=*/false);

    // One base address per subtile; the 8 rows this lane covers differ by a
    // compile-time constant (v*L floats = v*8192 bytes -> 24-bit imm offsets).
    const int rowBase = rOff0 + 8 * hlf;                 // C/D layout: m = vgpr + 8*(lane>=16)
    const size_t gBase = (size_t)(r0 + rowBase) * L + cc;
    const float* __restrict__ pA = con + gBase;
    float*       __restrict__ pX = X + gBase;
    const float* __restrict__ pB = &ldsT[cOff * (LT + 1) + rowBase];
    const int*   __restrict__ pD = &dR[rowBase];
    const int dBase = (r0 + rowBase) - cc;

    #pragma unroll
    for (int v = 0; v < 8; ++v) {
      float a = pA[(size_t)v * L];                       // direct element (imm-offset load)
      float b = pB[v];                                   // transposed element from LDS
      float s = 0.5f * (a + b);
      int d = dBase + v; d = (d < 0) ? -d : d;
      // Branchless mask: band AND (compat OR degen_r OR degen_c)
      int pm   = (compat[v] > 0.5f) ? 1 : 0;
      int keep = ((d >= 4) ? 1 : 0) & (pm | dc | pD[v]);
      pX[(size_t)v * L] = keep ? s : 0.0f;
    }
  }
}

// ---------- Matching: locally-dominant-edge rounds == sequential greedy ----------
__global__ void k_init_partner(int* __restrict__ partner) {
  int i = blockIdx.x * blockDim.x + threadIdx.x;
  if (i < L) partner[i] = -1;
}

// One block per row: best free column of a free row (strict max; ties -> lowest column).
__global__ __launch_bounds__(256) void k_rowbest(const float* __restrict__ X,
                                                 const int* __restrict__ partner,
                                                 int* __restrict__ bestc) {
  const int r = blockIdx.x;
  const int tid = threadIdx.x;
  if (partner[r] >= 0) { if (tid == 0) bestc[r] = -2; return; }  // uniform per block
  __shared__ float sv[256];
  __shared__ int   sc[256];
  float bv = 0.0f; int bc = -1;
  for (int c = tid; c < L; c += 256) {
    float v = X[(size_t)r * L + c];
    bool take = (v > bv) & (v > 0.0f) & (partner[c] < 0);
    bv = take ? v : bv;
    bc = take ? c : bc;
  }
  sv[tid] = bv; sc[tid] = bc;
  __syncthreads();
  for (int s = 128; s > 0; s >>= 1) {
    if (tid < s) {
      float v2 = sv[tid + s]; int c2 = sc[tid + s];
      bool take = (v2 > sv[tid]) | ((v2 == sv[tid]) & (c2 >= 0) & (c2 < sc[tid]));
      if (take) { sv[tid] = v2; sc[tid] = c2; }
    }
    __syncthreads();
  }
  if (tid == 0) bestc[r] = sc[0];
}

// Commit mutual-best pairs (r<c writes both sides; writer sets are disjoint).
__global__ void k_match(const int* __restrict__ bestc, int* __restrict__ partner) {
  int r = blockIdx.x * blockDim.x + threadIdx.x;
  if (r >= L) return;
  if (partner[r] >= 0) return;
  int c = bestc[r];
  if (c < 0) return;
  if (bestc[c] == r && r < c) { partner[r] = c; partner[c] = r; }
}

// ---------- Final: keep only matched entries (in place on d_out), float4 vectorized ----------
__global__ void k_apply(float4* __restrict__ X4, const int* __restrict__ partner) {
  size_t idx = (size_t)blockIdx.x * blockDim.x + threadIdx.x;
  if (idx >= (size_t)L * L / 4) return;
  size_t base = idx * 4;
  int r = (int)(base / L), c = (int)(base % L);   // L % 4 == 0 -> same row for all 4 lanes
  int p = partner[r];
  float4 v = X4[idx];
  v.x = (p == c    ) ? v.x : 0.0f;
  v.y = (p == c + 1) ? v.y : 0.0f;
  v.z = (p == c + 2) ? v.z : 0.0f;
  v.w = (p == c + 3) ? v.w : 0.0f;
  X4[idx] = v;
}

extern "C" void kernel_launch(void* const* d_in, const int* in_sizes, int n_in,
                              void* d_out, int out_size, void* d_ws, size_t ws_size,
                              hipStream_t stream) {
  const float* con  = (const float*)d_in[0];   // [1,1,L,L]
  const float* feat = (const float*)d_in[1];   // [1,8,L,L]
  float* X = (float*)d_out;                    // score matrix scratch == final output

  int* u       = (int*)d_ws;                   // 32 KB total workspace
  int* degen   = u + L;
  int* partner = degen + L;
  int* bestc   = partner + L;

  k_bases<<<(L + 255) / 256, 256, 0, stream>>>(feat, u, degen);
  k_build<<<(L / LT) * (L / LT), 128, 0, stream>>>(con, u, degen, X);
  k_init_partner<<<(L + 255) / 256, 256, 0, stream>>>(partner);
  for (int it = 0; it < NROUNDS; ++it) {
    k_rowbest<<<L, 256, 0, stream>>>(X, partner, bestc);
    k_match<<<(L + 255) / 256, 256, 0, stream>>>(bestc, partner);
  }
  k_apply<<<((size_t)L * L / 4 + 255) / 256, 256, 0, stream>>>((float4*)X, partner);
}
